// ConvRecLayer_7980049236010
// MI455X (gfx1250) — matrix operand order repaired and checked by reading, not compiled, not run on hardware
//
#include <hip/hip_runtime.h>
#include <hip/hip_bf16.h>
#include <cstdint>

// ---------------------------------------------------------------------------
// DynamicConv (fairseq-style) + LayerNorm + FFN(+residual) for gfx1250.
//   T=1024, B=8, C=1024, H=16, K=15, R=64, F=4096
//   k1: one WG per t, all 8 batch rows: proj -> softmax -> causal conv -> LN
//       (w_lin read once per WG: 8x less L2 traffic than per-(t,b) blocks)
//   k2: fc1_w (C,F) -> fc1T bf16 (F,C);  fc2_w (F,C) -> fc2T bf16 (C,F)
//   k3: out = relu(y@fc1+b1)@fc2 + b2 + y, fused; M=32 rows/WG so the 16MB
//       of bf16 weights are streamed only 256x (4GB L2 traffic) and every
//       B fragment feeds two WMMAs. h never leaves LDS.
// ---------------------------------------------------------------------------

#define TT 1024
#define BB 8
#define CC 1024
#define FF 4096
#define HH 16
#define KK 15
#define HK 240   // H*K

typedef __attribute__((ext_vector_type(16))) __bf16 v16bf;
typedef __attribute__((ext_vector_type(8)))  __bf16 v8bf;
typedef __attribute__((ext_vector_type(8)))  float  v8f;

// -------------------------------- kernel 1 ---------------------------------
// grid = T, block = 256 (8 waves; wave b handles LayerNorm of batch row b)
__global__ __launch_bounds__(256) void k_dynconv_ln(
    const float* __restrict__ x, const float* __restrict__ w_lin,
    const float* __restrict__ b_lin, const float* __restrict__ ln_g,
    const float* __restrict__ ln_b,
    float* __restrict__ y_f32, __bf16* __restrict__ y_b16)
{
  __shared__ float xcur[BB][CC];      // 32 KB: x[t,b,:], later reused as conv out
  __shared__ float wpj[BB][HK];       // 7.5 KB: projected weights (pre-softmax)
  __shared__ float wsm[BB][HH][KK];   // 7.5 KB: softmaxed taps

  const int t   = blockIdx.x;
  const int tid = threadIdx.x;

  // stage x[t, :, :] (8 rows) into LDS
  for (int i = tid; i < BB * CC; i += 256)
    xcur[i >> 10][i & (CC - 1)] = x[(size_t)t * BB * CC + i];
  __syncthreads();

  // projection for all 8 rows: each w_lin element loaded ONCE, 8 FMAs per load
  if (tid < HK) {
    float s[BB];
    #pragma unroll
    for (int b = 0; b < BB; ++b) s[b] = b_lin[tid];
    for (int c = 0; c < CC; ++c) {
      const float wv = w_lin[(size_t)c * HK + tid];   // coalesced over tid
      #pragma unroll
      for (int b = 0; b < BB; ++b) s[b] = fmaf(xcur[b][c], wv, s[b]);
    }
    #pragma unroll
    for (int b = 0; b < BB; ++b) wpj[b][tid] = s[b];
  }
  __syncthreads();

  // softmax over the K=15 taps, one thread per (b, h)
  if (tid < BB * HH) {
    const int b = tid >> 4, h = tid & 15;
    float m = -1e30f;
    #pragma unroll
    for (int k = 0; k < KK; ++k) m = fmaxf(m, wpj[b][h * KK + k]);
    float e[KK], s = 0.f;
    #pragma unroll
    for (int k = 0; k < KK; ++k) { e[k] = __expf(wpj[b][h * KK + k] - m); s += e[k]; }
    const float inv = 1.f / s;
    #pragma unroll
    for (int k = 0; k < KK; ++k) wsm[b][h][k] = e[k] * inv;
  }
  __syncthreads();

  // causal conv; window rows come from global (L2-resident, 15x reuse across t)
  for (int b = 0; b < BB; ++b) {
    for (int c = tid; c < CC; c += 256) {
      const int h = c >> 6;                       // R = 64 channels per head
      float acc = 0.f;
      #pragma unroll
      for (int k = 0; k < KK; ++k) {
        const int tr = t - (KK - 1) + k;
        const float xv = (tr >= 0) ? x[((size_t)tr * BB + b) * CC + c] : 0.f;
        acc = fmaf(wsm[b][h][k], xv, acc);
      }
      xcur[b][c] = acc;                           // overwrite with conv output
    }
  }
  __syncthreads();

  // LayerNorm: wave w normalizes batch row b = w (wave32 butterfly reduction)
  {
    const int b = tid >> 5, ln = tid & 31;
    float ps = 0.f, pq = 0.f;
    for (int c = ln; c < CC; c += 32) {
      const float v = xcur[b][c];
      ps += v; pq += v * v;
    }
    #pragma unroll
    for (int off = 16; off > 0; off >>= 1) {
      ps += __shfl_xor(ps, off, 32);
      pq += __shfl_xor(pq, off, 32);
    }
    const float mu   = ps / CC;
    const float rstd = rsqrtf(pq / CC - mu * mu + 1e-5f);
    for (int c = ln; c < CC; c += 32) {
      const float yv = (xcur[b][c] - mu) * rstd * ln_g[c] + ln_b[c];
      const size_t idx = ((size_t)t * BB + b) * CC + c;
      y_f32[idx] = yv;
      y_b16[idx] = (__bf16)yv;
    }
  }
}

// -------------------------------- kernel 2 ---------------------------------
__global__ __launch_bounds__(256) void k_cvt_w(
    const float* __restrict__ fc1_w, const float* __restrict__ fc2_w,
    __bf16* __restrict__ fc1T, __bf16* __restrict__ fc2T)
{
  const size_t n = (size_t)CC * FF;
  for (size_t i = (size_t)blockIdx.x * 256 + threadIdx.x; i < n;
       i += (size_t)gridDim.x * 256) {
    const size_t c = i / FF, f = i % FF;      // fc1_w (C,F) -> fc1T (F,C)
    fc1T[f * CC + c] = (__bf16)fc1_w[i];
    const size_t f2 = i / CC, c2 = i % CC;    // fc2_w (F,C) -> fc2T (C,F)
    fc2T[c2 * FF + f2] = (__bf16)fc2_w[i];
  }
}

// ----------------------------- WMMA fragments ------------------------------
// A (16x32 bf16): lane L: m=L&15, kb=kt+((L>>4)*16);
//   elems 0..15 -> K=kb..kb+15 (contiguous, matches B)
__device__ __forceinline__ v16bf load_frag_a(const __bf16* base, int stride,
                                             int kt, int lane)
{
  const int m  = lane & 15;
  const int kb = kt + ((lane >> 4) << 4);
  const __bf16* p = base + m * stride + kb;
  const v8bf lo = *(const v8bf*)p;
  const v8bf hi = *(const v8bf*)(p + 8);
  v16bf a;
  #pragma unroll
  for (int i = 0; i < 8; ++i) { a[i] = lo[i]; a[i + 8] = hi[i]; }
  return a;
}
// B (32x16 bf16): lane L: n=L&15, K = kt + (L>>4)*16 + i (contiguous 16 elems)
// -> one 32-byte load per lane from the transposed bf16 weights.

// -------------------------------- kernel 3 ---------------------------------
// grid = (T*B)/32 = 256 WGs; 8 waves; wave wv owns output cols [wv*128,+128)
// for TWO 16-row M-blocks (acc = 2x8 tiles = 128 VGPRs/lane).
__global__ __launch_bounds__(256) void k_ffn(
    const __bf16* __restrict__ yb, const float* __restrict__ yf,
    const __bf16* __restrict__ fc1T, const float* __restrict__ fc1_b,
    const __bf16* __restrict__ fc2T, const float* __restrict__ fc2_b,
    float* __restrict__ out)
{
  constexpr int FC = 128;   // F-chunk width
  constexpr int KC = 256;   // K-chunk of y staged in LDS
  __shared__ __attribute__((aligned(16))) __bf16 yA[32 * KC];  // 16 KB
  __shared__ __attribute__((aligned(16))) __bf16 hA[32 * FC];  //  8 KB

  const int tid  = threadIdx.x;
  const int wv   = tid >> 5;
  const int lane = tid & 31;
  const int row0 = blockIdx.x * 32;
  const int nn   = lane & 15;
  const int mb8  = (lane >> 4) << 3;   // D-tile row base for this lane half
  const int khi  = (lane >> 4) << 4;   // B-fragment K offset for this lane half

  const v8f vzero = {0.f, 0.f, 0.f, 0.f, 0.f, 0.f, 0.f, 0.f};
  v8f acc0[8], acc1[8];
  #pragma unroll
  for (int j = 0; j < 8; ++j) { acc0[j] = vzero; acc1[j] = vzero; }

  for (int fb = 0; fb < FF; fb += FC) {
    // ---- phase 1: h[32 x FC] = relu(y @ fc1 + b1); wave wv does 16 cols ----
    const int fcol = fb + wv * 16 + nn;
    const __bf16* bbase = fc1T + (size_t)fcol * CC + khi;
    if (fb + FC < FF)
      __builtin_prefetch(fc1T + (size_t)(fcol + FC) * CC, 0, 1);

    v8f h0 = vzero, h1 = vzero;
    for (int kb = 0; kb < CC; kb += KC) {
      // cooperatively stage y[row0..row0+31, kb..kb+KC) as bf16 in LDS
      #pragma unroll
      for (int it = 0; it < (32 * KC / 8) / 256; ++it) {
        const int e  = tid + it * 256;          // one v8bf per element
        const int m  = e >> 5;                  // KC/8 = 32 chunks per row
        const int c8 = e & 31;
        *(v8bf*)(yA + m * KC + c8 * 8) =
            *(const v8bf*)(yb + (size_t)(row0 + m) * CC + kb + c8 * 8);
      }
      __syncthreads();
      #pragma unroll
      for (int kt = 0; kt < KC; kt += 32) {
        const v16bf bf = *(const v16bf*)(bbase + kb + kt);
        const v16bf a0 = load_frag_a(yA,           KC, kt, lane);
        const v16bf a1 = load_frag_a(yA + 16 * KC, KC, kt, lane);
        h0 = __builtin_amdgcn_wmma_f32_16x16x32_bf16(false, a0, false, bf,
                                                     (short)0, h0, false, false);
        h1 = __builtin_amdgcn_wmma_f32_16x16x32_bf16(false, a1, false, bf,
                                                     (short)0, h1, false, false);
      }
      __syncthreads();   // before restaging yA
    }
    const float bias1 = fc1_b[fcol];
    #pragma unroll
    for (int r = 0; r < 8; ++r) {
      const float v0 = fmaxf(h0[r] + bias1, 0.f);
      const float v1 = fmaxf(h1[r] + bias1, 0.f);
      hA[(mb8 + r) * FC + wv * 16 + nn]        = (__bf16)v0;
      hA[(16 + mb8 + r) * FC + wv * 16 + nn]   = (__bf16)v1;
    }
    __syncthreads();

    // ---- phase 2: acc += h @ fc2 chunk; each B fragment feeds 2 WMMAs ----
    #pragma unroll
    for (int kc = 0; kc < 4; ++kc) {
      const v16bf a0 = load_frag_a(hA,           FC, kc * 32, lane);
      const v16bf a1 = load_frag_a(hA + 16 * FC, FC, kc * 32, lane);
      const int fk = fb + kc * 32 + khi;
      #pragma unroll
      for (int j = 0; j < 8; ++j) {
        const int c = wv * 128 + j * 16 + nn;
        const v16bf bf = *(const v16bf*)(fc2T + (size_t)c * FF + fk);
        acc0[j] = __builtin_amdgcn_wmma_f32_16x16x32_bf16(
            false, a0, false, bf, (short)0, acc0[j], false, false);
        acc1[j] = __builtin_amdgcn_wmma_f32_16x16x32_bf16(
            false, a1, false, bf, (short)0, acc1[j], false, false);
      }
    }
    __syncthreads();   // hA reused next chunk
  }

  // epilogue: + fc2 bias + residual y (post-LN), f32 out
  #pragma unroll
  for (int j = 0; j < 8; ++j) {
    const int c  = wv * 128 + j * 16 + nn;
    const float b2 = fc2_b[c];
    #pragma unroll
    for (int r = 0; r < 8; ++r) {
      const size_t r0 = (size_t)row0 + mb8 + r;
      const size_t r1 = r0 + 16;
      out[r0 * CC + c] = acc0[j][r] + b2 + yf[r0 * CC + c];
      out[r1 * CC + c] = acc1[j][r] + b2 + yf[r1 * CC + c];
    }
  }
}

// ------------------------------- launcher ----------------------------------
extern "C" void kernel_launch(void* const* d_in, const int* in_sizes, int n_in,
                              void* d_out, int out_size, void* d_ws, size_t ws_size,
                              hipStream_t stream) {
  const float* x     = (const float*)d_in[0];
  const float* w_lin = (const float*)d_in[1];
  const float* b_lin = (const float*)d_in[2];
  const float* ln_g  = (const float*)d_in[3];
  const float* ln_b  = (const float*)d_in[4];
  const float* fc1_w = (const float*)d_in[5];
  const float* fc1_b = (const float*)d_in[6];
  const float* fc2_w = (const float*)d_in[7];
  const float* fc2_b = (const float*)d_in[8];
  float* out = (float*)d_out;

  // workspace layout (64 MB total)
  char* ws = (char*)d_ws;
  float*  y_f32 = (float*)ws;                                   // 32 MB
  __bf16* y_b16 = (__bf16*)(ws + (size_t)32 * 1024 * 1024);     // 16 MB
  __bf16* fc1T  = (__bf16*)(ws + (size_t)48 * 1024 * 1024);     //  8 MB
  __bf16* fc2T  = (__bf16*)(ws + (size_t)56 * 1024 * 1024);     //  8 MB

  k_dynconv_ln<<<TT, 256, 0, stream>>>(x, w_lin, b_lin, ln_g, ln_b,
                                       y_f32, y_b16);
  k_cvt_w<<<4096, 256, 0, stream>>>(fc1_w, fc2_w, fc1T, fc2T);
  k_ffn<<<(TT * BB) / 32, 256, 0, stream>>>(y_b16, y_f32, fc1T, fc1_b,
                                            fc2T, fc2_b, out);
}
